// MHSA_29025388986873
// MI455X (gfx1250) — matrix operand order repaired and checked by reading, not compiled, not run on hardware
//
#include <hip/hip_runtime.h>

// ---------------------------------------------------------------------------
// Gated MHSA with pair bias (AlphaFold-style), fully fused on gfx1250 WMMA.
//   B=8, T=1024, C_M=256, H=8, D=32, C_Z=128
// Pipeline:
//   K0 convert : x -> f16, w_qkv -> w_qkvT f16, w_o -> w_oT f16
//   K1 pbias   : pair_bias[h][i][j] = bias_rep[i][j][:] . w_b[:,h] + b_b[h]
//   K2 qkv     : WMMA GEMM x_h @ w_qkv, scatter to q(scaled)/k/vT f16, sg=sigmoid(g)
//   K3 attn    : flash attention per (b,h), WMMA scores + WMMA P.V, gated, -> f16
//   K4 oproj   : WMMA GEMM attn_out_h @ w_oT -> d_out fp32
// ---------------------------------------------------------------------------

typedef __attribute__((ext_vector_type(16))) _Float16 v16h;
typedef __attribute__((ext_vector_type(8)))  _Float16 v8h;
typedef __attribute__((ext_vector_type(8)))  float    v8f;

#define HEADS 8
#define DHEAD 32
#define CM    256
#define CZ    128
#define BATCH 8
#define SEQ   1024
#define ROWS  (BATCH * SEQ)          // 8192
#define QKVN  (4 * HEADS * DHEAD)    // 1024

__device__ __forceinline__ v8f wmma_f16(v16h a, v16h b, v8f c) {
  // D = A(16x32 f16) x B(32x16 f16) + C(16x16 f32)
  return __builtin_amdgcn_wmma_f32_16x16x32_f16(false, a, false, b, (short)0, c,
                                                false, false);
}

// ---------------------------------------------------------------------------
// K0: precision conversion / transposition of small operands (L2 resident)
// ---------------------------------------------------------------------------
__global__ void convert_kernel(const float* __restrict__ x,
                               const float* __restrict__ w_qkv,
                               const float* __restrict__ w_o,
                               _Float16* __restrict__ x_h,
                               _Float16* __restrict__ wqkvT_h,
                               _Float16* __restrict__ woT_h) {
  const int stride = gridDim.x * blockDim.x;
  int tid = blockIdx.x * blockDim.x + threadIdx.x;
  for (int i = tid; i < ROWS * CM; i += stride) x_h[i] = (_Float16)x[i];
  for (int i = tid; i < QKVN * CM; i += stride) {
    int n = i / CM, k = i % CM;                 // wqkvT[n][k] = w_qkv[k][n]
    wqkvT_h[i] = (_Float16)w_qkv[(size_t)k * QKVN + n];
  }
  for (int i = tid; i < CM * CM; i += stride) {
    int n = i / CM, k = i % CM;                 // woT[n][k] = w_o[k][n]
    woT_h[i] = (_Float16)w_o[(size_t)k * CM + n];
  }
}

// ---------------------------------------------------------------------------
// K1: pair bias — the HBM-bound 512MB stream.  pb[h][i][j]
// ---------------------------------------------------------------------------
__global__ void pair_bias_kernel(const float* __restrict__ bias_rep,
                                 const float* __restrict__ w_b,
                                 const float* __restrict__ b_b,
                                 float* __restrict__ pb) {
  __shared__ float wbs[CZ * HEADS];
  __shared__ float bbs[HEADS];
  for (int i = threadIdx.x; i < CZ * HEADS; i += blockDim.x) wbs[i] = w_b[i];
  if (threadIdx.x < HEADS) bbs[threadIdx.x] = b_b[threadIdx.x];
  __syncthreads();

  const size_t ij = (size_t)blockIdx.x * blockDim.x + threadIdx.x;  // i*T + j
  const float4* src = (const float4*)(bias_rep + ij * CZ);

  float acc[HEADS];
#pragma unroll
  for (int h = 0; h < HEADS; ++h) acc[h] = bbs[h];

#pragma unroll 4
  for (int c4 = 0; c4 < CZ / 4; ++c4) {
    __builtin_prefetch(src + c4 + 16, 0, 0);    // global_prefetch_b8
    float4 v = src[c4];
    const int c = c4 * 4;
#pragma unroll
    for (int h = 0; h < HEADS; ++h) {
      acc[h] = fmaf(v.x, wbs[(c + 0) * HEADS + h],
               fmaf(v.y, wbs[(c + 1) * HEADS + h],
               fmaf(v.z, wbs[(c + 2) * HEADS + h],
               fmaf(v.w, wbs[(c + 3) * HEADS + h], acc[h]))));
    }
  }
#pragma unroll
  for (int h = 0; h < HEADS; ++h)
    pb[(size_t)h * SEQ * SEQ + ij] = acc[h];
}

// ---------------------------------------------------------------------------
// K2: fused QKV projection.  C[8192,1024] = x_h @ w_qkv, scattered:
//     col c -> d=c>>5, kk=(c>>3)&3, h=c&7   (reshape 'b t (d k h)')
//     q scaled by 1/sqrt(D); v stored transposed [bh][d][t]; g -> sigmoid
// ---------------------------------------------------------------------------
__global__ void qkv_kernel(const _Float16* __restrict__ x_h,
                           const _Float16* __restrict__ wqkvT_h,
                           _Float16* __restrict__ q_h,
                           _Float16* __restrict__ k_h,
                           _Float16* __restrict__ vT_h,
                           float* __restrict__ sg) {
  const int lane = threadIdx.x & 31;
  const int wv   = threadIdx.x >> 5;
  const int tile = blockIdx.x * 8 + wv;          // 32768 tiles
  const int tn   = tile & 63;                    // N tiles: 1024/16
  const int tm   = tile >> 6;                    // M tiles: 8192/16

  const int arow = tm * 16 + (lane & 15);
  const int aseg = (lane >> 4) * 8;              // A layout K sub-segment
  const int bcol = tn * 16 + (lane & 15);        // B layout: lane = column
  const int bseg = (lane >> 4) * 16;             // B layout K half

  v8f c = {};
#pragma unroll
  for (int k = 0; k < CM; k += 32) {
    v16h a = *(const v16h*)(x_h + (size_t)arow * CM + k + bseg);
    v16h b = *(const v16h*)(wqkvT_h + (size_t)bcol * CM + k + bseg);
    c = wmma_f16(a, b, c);
  }

  // scatter C layout: vgpr r, lane n -> (m = r + 8*(lane>>4), col = tn*16 + n)
#pragma unroll
  for (int r = 0; r < 8; ++r) {
    const int m    = r + 8 * (lane >> 4);
    const int rowg = tm * 16 + m;
    const int b_   = rowg >> 10;
    const int t_   = rowg & (SEQ - 1);
    const int colg = tn * 16 + (lane & 15);
    const int d    = colg >> 5;
    const int kk   = (colg >> 3) & 3;
    const int h    = colg & 7;
    const size_t bh = (size_t)b_ * HEADS + h;
    const float val = c[r];
    if (kk == 0)      q_h[(bh * SEQ + t_) * DHEAD + d] = (_Float16)(val * 0.17677669529663687f);
    else if (kk == 1) k_h[(bh * SEQ + t_) * DHEAD + d] = (_Float16)val;
    else if (kk == 2) vT_h[(bh * DHEAD + d) * SEQ + t_] = (_Float16)val;
    else              sg[(bh * SEQ + t_) * DHEAD + d] = 1.0f / (1.0f + __expf(-val));
  }
}

// ---------------------------------------------------------------------------
// K3: flash attention.  Block = 128 thr (4 waves), wave owns 16 query rows.
//     Per 32-key tile: 2 score WMMAs (K=32=DHEAD), online softmax,
//     P transposed through LDS, 2 P.V WMMAs with fp32 accumulators.
// ---------------------------------------------------------------------------
__global__ void attn_kernel(const _Float16* __restrict__ q_h,
                            const _Float16* __restrict__ k_h,
                            const _Float16* __restrict__ vT_h,
                            const float* __restrict__ sg,
                            const float* __restrict__ pb,
                            _Float16* __restrict__ attn_out_h) {
  const int lane = threadIdx.x & 31;
  const int wv   = threadIdx.x >> 5;
  const int h    = blockIdx.y;
  const int b_   = blockIdx.z;
  const int row0 = blockIdx.x * 64 + wv * 16;
  const size_t bh = (size_t)b_ * HEADS + h;

  const _Float16* qb  = q_h  + bh * SEQ * DHEAD;
  const _Float16* kb  = k_h  + bh * SEQ * DHEAD;
  const _Float16* vtb = vT_h + bh * DHEAD * SEQ;
  const float*    pbb = pb   + (size_t)h * SEQ * SEQ;

  __shared__ __align__(32) _Float16 lds_p[4][16][32];

  const int mlane = lane & 15;
  const int aseg  = (lane >> 4) * 8;
  const int bseg  = (lane >> 4) * 16;
  const int mbase = 8 * (lane >> 4);

  // A operand: q tile, loaded once (K = DHEAD = 32 fits one WMMA)
  v16h qa;
  {
    qa = *(const v16h*)(qb + (size_t)(row0 + mlane) * DHEAD + bseg);
  }

  float row_max[8], row_sum[8];
  v8f oA = {}, oB = {};
#pragma unroll
  for (int r = 0; r < 8; ++r) { row_max[r] = -1e30f; row_sum[r] = 0.0f; }

  for (int jt = 0; jt < SEQ / 32; ++jt) {
    const int kt = jt * 32;

    // B operands: two 16-key column groups, K = 32 dims (contiguous rows)
    v16h kb0 = *(const v16h*)(kb + (size_t)(kt + mlane)      * DHEAD + bseg);
    v16h kb1 = *(const v16h*)(kb + (size_t)(kt + 16 + mlane) * DHEAD + bseg);
    v8f z = {};
    v8f s0 = wmma_f16(qa, kb0, z);
    v8f s1 = wmma_f16(qa, kb1, z);

    // add pair bias (C layout: vgpr r -> row mbase+r, lane -> column)
#pragma unroll
    for (int r = 0; r < 8; ++r) {
      const size_t rg = (size_t)(row0 + mbase + r) * SEQ;
      s0[r] += pbb[rg + kt + mlane];
      s1[r] += pbb[rg + kt + mlane + 16];
    }

    // online softmax per row (row lives in 16 lanes of one half-wave)
#pragma unroll
    for (int r = 0; r < 8; ++r) {
      float mx = fmaxf(s0[r], s1[r]);
      mx = fmaxf(mx, __shfl_xor(mx, 1, 32));
      mx = fmaxf(mx, __shfl_xor(mx, 2, 32));
      mx = fmaxf(mx, __shfl_xor(mx, 4, 32));
      mx = fmaxf(mx, __shfl_xor(mx, 8, 32));
      const float mnew  = fmaxf(row_max[r], mx);
      const float alpha = __expf(row_max[r] - mnew);
      row_max[r] = mnew;
      const float p0 = __expf(s0[r] - mnew);
      const float p1 = __expf(s1[r] - mnew);
      float sum = p0 + p1;
      sum += __shfl_xor(sum, 1, 32);
      sum += __shfl_xor(sum, 2, 32);
      sum += __shfl_xor(sum, 4, 32);
      sum += __shfl_xor(sum, 8, 32);
      row_sum[r] = row_sum[r] * alpha + sum;
      oA[r] *= alpha;
      oB[r] *= alpha;
      lds_p[wv][mbase + r][mlane]      = (_Float16)p0;
      lds_p[wv][mbase + r][mlane + 16] = (_Float16)p1;
    }
    __syncthreads();   // uniform across all 4 waves; publishes P transpose

    // reload P in A layout (16 queries x 32 keys)
    v16h pa;
    {
      pa = *(const v16h*)(&lds_p[wv][mlane][bseg]);
    }

    // B operands from transposed V: contiguous 16-key runs per dim
    v16h vb0 = *(const v16h*)(vtb + (size_t)mlane        * SEQ + kt + bseg);
    v16h vb1 = *(const v16h*)(vtb + (size_t)(mlane + 16) * SEQ + kt + bseg);
    oA = wmma_f16(pa, vb0, oA);
    oB = wmma_f16(pa, vb1, oB);
    __syncthreads();
  }

  // epilogue: normalize, gate by sigmoid(g), store f16 in [b,t,(h d)] layout
#pragma unroll
  for (int r = 0; r < 8; ++r) {
    const int rg  = row0 + mbase + r;
    const float inv = 1.0f / row_sum[r];
    const int d0 = mlane;
    const float g0 = sg[(bh * SEQ + rg) * DHEAD + d0];
    const float g1 = sg[(bh * SEQ + rg) * DHEAD + d0 + 16];
    const size_t ob = ((size_t)b_ * SEQ + rg) * CM + h * DHEAD;
    attn_out_h[ob + d0]      = (_Float16)(oA[r] * inv * g0);
    attn_out_h[ob + d0 + 16] = (_Float16)(oB[r] * inv * g1);
  }
}

// ---------------------------------------------------------------------------
// K4: output projection.  d_out[8192,256] = attn_out_h @ w_o  (fp32 out)
// ---------------------------------------------------------------------------
__global__ void oproj_kernel(const _Float16* __restrict__ ao_h,
                             const _Float16* __restrict__ woT_h,
                             float* __restrict__ out) {
  const int lane = threadIdx.x & 31;
  const int wv   = threadIdx.x >> 5;
  const int tile = blockIdx.x * 8 + wv;          // 8192 tiles
  const int tn   = tile & 15;                    // 256/16
  const int tm   = tile >> 4;                    // 8192/16

  const int arow = tm * 16 + (lane & 15);
  const int aseg = (lane >> 4) * 8;
  const int bcol = tn * 16 + (lane & 15);
  const int bseg = (lane >> 4) * 16;

  v8f c = {};
#pragma unroll
  for (int k = 0; k < CM; k += 32) {
    v16h a = *(const v16h*)(ao_h + (size_t)arow * CM + k + bseg);
    v16h b = *(const v16h*)(woT_h + (size_t)bcol * CM + k + bseg);
    c = wmma_f16(a, b, c);
  }
#pragma unroll
  for (int r = 0; r < 8; ++r) {
    const int rowg = tm * 16 + r + 8 * (lane >> 4);
    const int colg = tn * 16 + (lane & 15);
    out[(size_t)rowg * CM + colg] = c[r];
  }
}

// ---------------------------------------------------------------------------
// Launch
// ---------------------------------------------------------------------------
extern "C" void kernel_launch(void* const* d_in, const int* in_sizes, int n_in,
                              void* d_out, int out_size, void* d_ws, size_t ws_size,
                              hipStream_t stream) {
  const float* x        = (const float*)d_in[0];
  const float* bias_rep = (const float*)d_in[1];
  const float* w_qkv    = (const float*)d_in[2];
  const float* w_o      = (const float*)d_in[3];
  const float* w_b      = (const float*)d_in[4];
  const float* b_b      = (const float*)d_in[5];
  float* out            = (float*)d_out;

  // workspace carve-up (256B aligned)
  char* ws = (char*)d_ws;
  size_t off = 0;
  auto alloc = [&](size_t bytes) { void* p = ws + off; off = (off + bytes + 255) & ~(size_t)255; return p; };
  float*    pb      = (float*)   alloc((size_t)HEADS * SEQ * SEQ * 4);      // 32 MB
  _Float16* q_h     = (_Float16*)alloc((size_t)BATCH * HEADS * SEQ * DHEAD * 2);
  _Float16* k_h     = (_Float16*)alloc((size_t)BATCH * HEADS * SEQ * DHEAD * 2);
  _Float16* vT_h    = (_Float16*)alloc((size_t)BATCH * HEADS * SEQ * DHEAD * 2);
  float*    sg      = (float*)   alloc((size_t)BATCH * HEADS * SEQ * DHEAD * 4);
  _Float16* ao_h    = (_Float16*)alloc((size_t)ROWS * CM * 2);
  _Float16* x_h     = (_Float16*)alloc((size_t)ROWS * CM * 2);
  _Float16* wqkvT_h = (_Float16*)alloc((size_t)QKVN * CM * 2);
  _Float16* woT_h   = (_Float16*)alloc((size_t)CM * CM * 2);

  convert_kernel<<<2048, 256, 0, stream>>>(x, w_qkv, w_o, x_h, wqkvT_h, woT_h);
  pair_bias_kernel<<<(SEQ * SEQ) / 256, 256, 0, stream>>>(bias_rep, w_b, b_b, pb);
  qkv_kernel<<<(512 * 64) / 8, 256, 0, stream>>>(x_h, wqkvT_h, q_h, k_h, vT_h, sg);
  attn_kernel<<<dim3(SEQ / 64, HEADS, BATCH), 128, 0, stream>>>(q_h, k_h, vT_h, sg, pb, ao_h);
  oproj_kernel<<<(512 * 16) / 8, 256, 0, stream>>>(ao_h, woT_h, out);
}
